// EMSA_61546881352021
// MI455X (gfx1250) — matrix operand order repaired and checked by reading, not compiled, not run on hardware
//
#include <hip/hip_runtime.h>

typedef __attribute__((ext_vector_type(16))) _Float16 v16h;
typedef __attribute__((ext_vector_type(8)))  _Float16 v8h;
typedef __attribute__((ext_vector_type(4)))  _Float16 v4h;
typedef __attribute__((ext_vector_type(8)))  float    v8f;

namespace {
constexpr int BB  = 16;
constexpr int HW  = 48;
constexpr int DIM = 512;
constexpr int NH  = 8;
constexpr int DK  = 64;
constexpr int NQ  = HW * HW;     // 2304
constexpr int NKV = 256;         // 16x16 after stride-3 conv
constexpr int BH  = BB * NH;     // 128
constexpr float EPSV = 1e-5f;
}

// ---------------------------------------------------------------------------
// WMMA fragment loaders (gfx1250 16x16x32 f16; this code's contiguous k order, same in A and B)
// A (16x32, MxK): lanes 0-15 hold row M=lane, K in [0,16);
//                 lanes 16-31 hold K in [16,32).
// B (32x16, KxN): stored as B^T row-major; lane n<16 holds col n, K=[0,16);
//                 lanes 16-31 hold K=[16,32). 16 contiguous halfs per lane.
// ---------------------------------------------------------------------------
__device__ inline v16h load_fragA(const _Float16* base, int ld) {
  int lane = threadIdx.x & 31;
  const _Float16* p = base + (size_t)(lane & 15) * ld + ((lane >> 4) << 4);
  v8h lo = *(const v8h*)p;
  v8h hi = *(const v8h*)(p + 8);
  return __builtin_shufflevector(lo, hi, 0,1,2,3,4,5,6,7,8,9,10,11,12,13,14,15);
}

__device__ inline v16h load_fragB(const _Float16* baseT, int ld) {
  int lane = threadIdx.x & 31;
  const _Float16* p = baseT + (size_t)(lane & 15) * ld + ((lane >> 4) << 4);
  return *(const v16h*)p;
}

// ---------------------------------------------------------------------------
// Tiled WMMA GEMM: C[M,N] = A[M,K] * Bt[N,K]^T (+bias), per-MODE epilogue.
// Block = 256 threads = 8 waves arranged (8/WGN) x WGN; each wave computes a
// 32(M) x 64(N) tile: 2 A-frags x 4 B-frags -> 8 WMMAs per 12 VMEM loads per
// K=32 step (vs 4 WMMAs per 10 loads with a 16x64 tile).
//   MODE 0: Q proj   -> f16 [(b*8+h), q, dk]
//   MODE 1: K proj   -> f16 [(b*8+h), nk, dk]
//   MODE 2: V proj   -> f16 [(b*8+h), dv, nk]   (transposed for PV GEMM)
//   MODE 3: scores   -> f16 [(bh), q, k] * 1/sqrt(dk); per-bh A/Bt (blockIdx.z)
//   MODE 4: P*V      -> f16 attn_out[b*2304+q, h*64+dv] = (acc - cs/256)*inv_s
//   MODE 5: out proj -> f32 row-major d_out
// ---------------------------------------------------------------------------
template <int MODE, int WGN>
__global__ void __launch_bounds__(256)
gemm16(const _Float16* __restrict__ A, const _Float16* __restrict__ Bt,
       const float* __restrict__ bias, void* __restrict__ out,
       const float* __restrict__ colsumV, const float* __restrict__ inv_s,
       int K, int lda, int ldb) {
  const int w  = threadIdx.x >> 5;
  const int wm = w / WGN, wn = w % WGN;
  const int m0 = (blockIdx.y * (8 / WGN) + wm) * 32;
  const int n0 = (blockIdx.x * WGN + wn) * 64;
  const int bh = blockIdx.z;

  if constexpr (MODE == 3) { A += (size_t)bh * NQ * DK;  Bt += (size_t)bh * NKV * DK; }
  if constexpr (MODE == 4) { A += (size_t)bh * NQ * NKV; Bt += (size_t)bh * DK * NKV; }

  const _Float16* Ap = A + (size_t)m0 * lda;
  v8f acc[2][4] = {};
  for (int kk = 0; kk < K; kk += 32) {
    v16h a0 = load_fragA(Ap + kk, lda);
    v16h a1 = load_fragA(Ap + (size_t)16 * lda + kk, lda);
#pragma unroll
    for (int j = 0; j < 4; ++j) {
      v16h b = load_fragB(Bt + (size_t)(n0 + 16 * j) * ldb + kk, ldb);
      acc[0][j] = __builtin_amdgcn_wmma_f32_16x16x32_f16(
          false, a0, false, b, (short)0, acc[0][j], false, false);
      acc[1][j] = __builtin_amdgcn_wmma_f32_16x16x32_f16(
          false, a1, false, b, (short)0, acc[1][j], false, false);
    }
  }

  // C/D layout: VGPR r -> M = m0 + 16i + r + (lane>=16 ? 8 : 0),
  //             N = n0 + 16j + (lane&15)
  const int lane = threadIdx.x & 31;
  const int rb = (lane >> 4) << 3;
  const int nn = lane & 15;
#pragma unroll
  for (int i = 0; i < 2; ++i) {
#pragma unroll
    for (int j = 0; j < 4; ++j) {
      const int n = n0 + j * 16 + nn;
#pragma unroll
      for (int r = 0; r < 8; ++r) {
        const int m = m0 + i * 16 + rb + r;
        float val = acc[i][j][r];
        if constexpr (MODE == 0) {
          size_t idx = (((size_t)(m / NQ) * NH + (n >> 6)) * NQ + (m % NQ)) * DK + (n & 63);
          ((_Float16*)out)[idx] = (_Float16)(val + bias[n]);
        } else if constexpr (MODE == 1) {
          size_t idx = (((size_t)(m >> 8) * NH + (n >> 6)) * NKV + (m & 255)) * DK + (n & 63);
          ((_Float16*)out)[idx] = (_Float16)(val + bias[n]);
        } else if constexpr (MODE == 2) {
          size_t idx = (((size_t)(m >> 8) * NH + (n >> 6)) * DK + (n & 63)) * NKV + (m & 255);
          ((_Float16*)out)[idx] = (_Float16)(val + bias[n]);
        } else if constexpr (MODE == 3) {
          size_t idx = ((size_t)bh * NQ + m) * NKV + n;
          ((_Float16*)out)[idx] = (_Float16)(val * 0.125f);  // 1/sqrt(64)
        } else if constexpr (MODE == 4) {
          float cs = colsumV[bh * DK + n];
          float v  = (val - (1.0f / NKV) * cs) * inv_s[bh];
          size_t idx = ((size_t)((bh >> 3) * NQ + m)) * DIM + (bh & 7) * DK + n;
          ((_Float16*)out)[idx] = (_Float16)v;
        } else {  // MODE 5
          ((float*)out)[(size_t)m * DIM + n] = val + bias[n];
        }
      }
    }
  }
}

// --------------------------- f32 -> f16 convert ----------------------------
__global__ void __launch_bounds__(256)
k_cvt(const float* __restrict__ x, _Float16* __restrict__ y) {
  size_t i = (size_t)blockIdx.x * 256 + threadIdx.x;  // one float4 per thread
  float4 v = ((const float4*)x)[i];
  v4h o; o[0] = (_Float16)v.x; o[1] = (_Float16)v.y;
         o[2] = (_Float16)v.z; o[3] = (_Float16)v.w;
  ((v4h*)y)[i] = o;
}

// --------------- weight transpose + convert: Wt[n,k] = W[k,n] --------------
__global__ void __launch_bounds__(256)
k_transw(const float* __restrict__ W0, const float* __restrict__ W1,
         const float* __restrict__ W2, const float* __restrict__ W3,
         _Float16* __restrict__ Wt) {
  __shared__ float tile[32][33];
  const int z = blockIdx.z;
  const float* W = (z == 0) ? W0 : (z == 1) ? W1 : (z == 2) ? W2 : W3;
  _Float16* dst = Wt + (size_t)z * DIM * DIM;
  const int x = blockIdx.x * 32 + threadIdx.x;  // col n of W
  const int y0 = blockIdx.y * 32;               // row k of W
#pragma unroll
  for (int i = 0; i < 4; ++i)
    tile[threadIdx.y + i * 8][threadIdx.x] = W[(size_t)(y0 + threadIdx.y + i * 8) * DIM + x];
  __syncthreads();
#pragma unroll
  for (int i = 0; i < 4; ++i)
    dst[(size_t)(blockIdx.x * 32 + threadIdx.y + i * 8) * DIM + y0 + threadIdx.x] =
        (_Float16)tile[threadIdx.x][threadIdx.y + i * 8];
}

// -------- depthwise 4x4 stride-3 pad-1 conv: queries -> xr (b,nk,512) ------
__global__ void __launch_bounds__(256)
k_conv(const float* __restrict__ q, const float* __restrict__ w,
       const float* __restrict__ b, float* __restrict__ xr) {
  int idx = blockIdx.x * 256 + threadIdx.x;   // 16*256*512 total
  int c  = idx & 511;
  int nk = (idx >> 9) & 255;
  int bb = idx >> 17;
  int oy = nk >> 4, ox = nk & 15;
  float acc = b[c];
#pragma unroll
  for (int ky = 0; ky < 4; ++ky) {
    int y = oy * 3 - 1 + ky;
    if ((unsigned)y >= (unsigned)HW) continue;
#pragma unroll
    for (int kx = 0; kx < 4; ++kx) {
      int x = ox * 3 - 1 + kx;
      if ((unsigned)x >= (unsigned)HW) continue;
      acc += q[((size_t)bb * NQ + y * HW + x) * DIM + c] * w[c * 16 + ky * 4 + kx];
    }
  }
  xr[idx] = acc;
}

// --------------- LayerNorm over 512 channels -> f16 rows -------------------
__global__ void __launch_bounds__(256)
k_ln(const float* __restrict__ xr, const float* __restrict__ g,
     const float* __restrict__ bta, _Float16* __restrict__ xln) {
  __shared__ float s1[256], s2[256];
  const int row = blockIdx.x;
  const int t = threadIdx.x;
  float a = xr[(size_t)row * DIM + t];
  float b = xr[(size_t)row * DIM + t + 256];
  s1[t] = a + b;
  s2[t] = a * a + b * b;
  __syncthreads();
  for (int off = 128; off > 0; off >>= 1) {
    if (t < off) { s1[t] += s1[t + off]; s2[t] += s2[t + off]; }
    __syncthreads();
  }
  float mu  = s1[0] * (1.0f / DIM);
  float var = s2[0] * (1.0f / DIM) - mu * mu;
  float inv = rsqrtf(var + EPSV);
  xln[(size_t)row * DIM + t]       = (_Float16)((a - mu) * inv * g[t] + bta[t]);
  xln[(size_t)row * DIM + t + 256] = (_Float16)((b - mu) * inv * g[t + 256] + bta[t + 256]);
}

// ---- head mix (tw,tb) + softmax over nk + sum(p^2) atomic, in place -------
// one wave32 per (b,q) row; each lane owns 8 contiguous k per head.
__global__ void __launch_bounds__(256)
k_mix(_Float16* __restrict__ S, const float* __restrict__ tw,
      const float* __restrict__ tb, float* __restrict__ ssq) {
  const int gwave = (blockIdx.x * 256 + threadIdx.x) >> 5;  // b*2304+q
  const int lane = threadIdx.x & 31;
  const int b = gwave / NQ, q = gwave % NQ;

  float s[NH][8];
#pragma unroll
  for (int h = 0; h < NH; ++h) {
    const v8h* p = (const v8h*)(S + (((size_t)(b * NH + h) * NQ + q) * NKV) + lane * 8);
    v8h x = *p;
#pragma unroll
    for (int j = 0; j < 8; ++j) s[h][j] = (float)x[j];
  }

  for (int o = 0; o < NH; ++o) {
    float m[8];
#pragma unroll
    for (int j = 0; j < 8; ++j) {
      float acc = tb[o];
#pragma unroll
      for (int i = 0; i < NH; ++i) acc += tw[o * NH + i] * s[i][j];
      m[j] = acc;
    }
    float mx = m[0];
#pragma unroll
    for (int j = 1; j < 8; ++j) mx = fmaxf(mx, m[j]);
    for (int off = 16; off > 0; off >>= 1) mx = fmaxf(mx, __shfl_xor(mx, off, 32));
    float sum = 0.f;
#pragma unroll
    for (int j = 0; j < 8; ++j) { m[j] = __expf(m[j] - mx); sum += m[j]; }
    for (int off = 16; off > 0; off >>= 1) sum += __shfl_xor(sum, off, 32);
    float inv = 1.0f / sum, sq = 0.f;
    v8h outv;
#pragma unroll
    for (int j = 0; j < 8; ++j) {
      float p = m[j] * inv;
      sq += p * p;
      outv[j] = (_Float16)p;
    }
    for (int off = 16; off > 0; off >>= 1) sq += __shfl_xor(sq, off, 32);
    if (lane == 0) atomicAdd(&ssq[b * NH + o], sq);
    *(v8h*)(S + (((size_t)(b * NH + o) * NQ + q) * NKV) + lane * 8) = outv;
  }
}

// --------------- column sums of V: one wave per (bh,dv) row of Vt ----------
__global__ void __launch_bounds__(256)
k_colsum(const _Float16* __restrict__ Vt, float* __restrict__ colsumV) {
  const int wid = (blockIdx.x * 256 + threadIdx.x) >> 5;  // bh*64+dv, 8192 rows
  const int lane = threadIdx.x & 31;
  v8h x = *(const v8h*)(Vt + (size_t)wid * NKV + lane * 8);
  float sum = 0.f;
#pragma unroll
  for (int j = 0; j < 8; ++j) sum += (float)x[j];
  for (int off = 16; off > 0; off >>= 1) sum += __shfl_xor(sum, off, 32);
  if (lane == 0) colsumV[wid] = sum;
}

__global__ void k_zero(float* __restrict__ ssq) { ssq[threadIdx.x] = 0.f; }

__global__ void k_stats(const float* __restrict__ ssq, float* __restrict__ inv_s) {
  int t = threadIdx.x;  // 128
  float m  = 1.0f / NKV;                 // softmax rows sum to 1 -> exact mean
  float va = ssq[t] / ((float)NQ * NKV) - m * m;
  inv_s[t] = rsqrtf(va + EPSV);
}

// ---------------------------------------------------------------------------
extern "C" void kernel_launch(void* const* d_in, const int* in_sizes, int n_in,
                              void* d_out, int out_size, void* d_ws, size_t ws_size,
                              hipStream_t stream) {
  const float* queries = (const float*)d_in[0];
  const float* Wq = (const float*)d_in[1];  const float* bq = (const float*)d_in[2];
  const float* Wk = (const float*)d_in[3];  const float* bk = (const float*)d_in[4];
  const float* Wv = (const float*)d_in[5];  const float* bv = (const float*)d_in[6];
  const float* Wo = (const float*)d_in[7];  const float* bo = (const float*)d_in[8];
  const float* sr_w = (const float*)d_in[9];  const float* sr_b = (const float*)d_in[10];
  const float* ln_g = (const float*)d_in[11]; const float* ln_b = (const float*)d_in[12];
  const float* tw = (const float*)d_in[13];   const float* tb = (const float*)d_in[14];

  char* ws = (char*)d_ws;
  size_t off = 0;
  auto alloc = [&](size_t bytes) {
    void* p = ws + off;
    off += (bytes + 255) & ~(size_t)255;
    return p;
  };
  const size_t MQ = (size_t)BB * NQ;          // 36864 rows
  _Float16* Xq16 = (_Float16*)alloc(MQ * DIM * 2);        // reused as attn_out (A2)
  _Float16* Qf16 = (_Float16*)alloc(MQ * DIM * 2);
  _Float16* Sbuf = (_Float16*)alloc((size_t)BH * NQ * NKV * 2);  // scores, then P in place
  _Float16* Wt   = (_Float16*)alloc((size_t)4 * DIM * DIM * 2);
  float*    xr   = (float*)alloc((size_t)BB * NKV * DIM * 4);
  _Float16* xln  = (_Float16*)alloc((size_t)BB * NKV * DIM * 2);
  _Float16* Kf16 = (_Float16*)alloc((size_t)BH * NKV * DK * 2);
  _Float16* Vt   = (_Float16*)alloc((size_t)BH * DK * NKV * 2);
  float* colsumV = (float*)alloc((size_t)BH * DK * 4);
  float* ssq     = (float*)alloc(BH * 4);
  float* inv_s   = (float*)alloc(BH * 4);
  _Float16* A2 = Xq16;  // Xq16 dead after Q projection

  k_zero<<<1, BH, 0, stream>>>(ssq);
  k_cvt<<<(int)(MQ * DIM / 4 / 256), 256, 0, stream>>>(queries, Xq16);
  k_transw<<<dim3(16, 16, 4), dim3(32, 8), 0, stream>>>(Wq, Wk, Wv, Wo, Wt);

  // Q projection -> (b,h,q,dk) f16         (M=36864, N=512, K=512)
  gemm16<0, 2><<<dim3(4, 288, 1), 256, 0, stream>>>(
      Xq16, Wt, bq, Qf16, nullptr, nullptr, DIM, DIM, DIM);

  // spatial reduction conv + LayerNorm
  k_conv<<<(BB * NKV * DIM) / 256, 256, 0, stream>>>(queries, sr_w, sr_b, xr);
  k_ln<<<BB * NKV, 256, 0, stream>>>(xr, ln_g, ln_b, xln);

  // K, V projections                        (M=4096, N=512, K=512)
  gemm16<1, 2><<<dim3(4, 32, 1), 256, 0, stream>>>(
      xln, Wt + (size_t)1 * DIM * DIM, bk, Kf16, nullptr, nullptr, DIM, DIM, DIM);
  gemm16<2, 2><<<dim3(4, 32, 1), 256, 0, stream>>>(
      xln, Wt + (size_t)2 * DIM * DIM, bv, Vt, nullptr, nullptr, DIM, DIM, DIM);
  k_colsum<<<(BH * DK * 32) / 256, 256, 0, stream>>>(Vt, colsumV);

  // scores S = Q K^T / 8 per (b,h)          (M=2304, N=256, K=64) x 128
  gemm16<3, 2><<<dim3(2, 18, BH), 256, 0, stream>>>(
      Qf16, Kf16, nullptr, Sbuf, nullptr, nullptr, DK, DK, DK);

  // head mix + softmax + sum(p^2), in place
  k_mix<<<(int)(MQ * 32 / 256), 256, 0, stream>>>(Sbuf, tw, tb, ssq);
  k_stats<<<1, BH, 0, stream>>>(ssq, inv_s);

  // O = P V with fused InstanceNorm epilogue (M=2304, N=64, K=256) x 128
  gemm16<4, 1><<<dim3(1, 9, BH), 256, 0, stream>>>(
      Sbuf, Vt, nullptr, A2, colsumV, inv_s, NKV, NKV, NKV);

  // output projection -> f32 d_out          (M=36864, N=512, K=512)
  gemm16<5, 2><<<dim3(4, 288, 1), 256, 0, stream>>>(
      A2, Wt + (size_t)3 * DIM * DIM, bo, d_out, nullptr, nullptr, DIM, DIM, DIM);
}
